// AdaptiveSoftmaxProbe_37469294691089
// MI455X (gfx1250) — matrix operand order repaired and checked by reading, not compiled, not run on hardware
//
#include <hip/hip_runtime.h>
#include <hip/hip_bf16.h>

// ---------------------------------------------------------------------------
// Adaptive softmax (fused GEMM + online logsumexp) for MI455X / gfx1250.
//   head : logits = x(4096x1024) @ W_head^T(1002x1024)   -> LSE + cols {tgt,1000,1001}
//   tail0: p0 = x @ W0a^T (4096x256, bf16), logits = p0 @ W0b^T(9000x256) -> LSE + tgt
//   tail1: p1 = x @ W1a^T (4096x64,  bf16), logits = p1 @ W1b^T(40257x64) -> LSE + tgt
// Never materializes the 4096x50257 log-prob matrix (saves ~1.6GB HBM traffic).
// GEMMs: v_wmma_f32_16x16x32_bf16 (fp32 accumulate); all softmax math fp32.
// A panels staged to LDS with global_load_async_to_lds_b128 (ASYNCcnt).
// Per-lane online LSE in the hot loop (no shuffles per tile); cross-lane /
// cross-wave merge happens once at the end.
// ---------------------------------------------------------------------------

typedef unsigned short ushort_t;
typedef __attribute__((ext_vector_type(16))) __bf16 v16bf;
typedef __attribute__((ext_vector_type(8)))  float  v8f;

#define N_TOKENS 4096
#define D_MODEL  1024

#define NH_REAL 1002
#define NH_PAD  1008
#define N0_REAL 9000
#define N0_PAD  9008
#define N1_REAL 40257
#define N1_PAD  40272

#define LSE_NEG (-1.0e30f)     // finite sentinel: exp(LSE_NEG - m) == 0, no NaN paths

#define USE_ASYNC_STAGE 1

union Frag32 { uint4 u[2]; v16bf v; };
__device__ __forceinline__ v16bf make_frag(uint4 lo, uint4 hi) {
  Frag32 f; f.u[0] = lo; f.u[1] = hi; return f.v;
}

// round-to-nearest-even f32 -> bf16 (raw bits)
__device__ __forceinline__ ushort_t f2bf(float f) {
  unsigned int u = __float_as_uint(f);
  unsigned int r = (u + 0x7FFFu + ((u >> 16) & 1u)) >> 16;
  return (ushort_t)r;
}

// merge two (max, sumexp) partials
__device__ __forceinline__ void lse_merge(float& m, float& s, float mo, float so) {
  float nm = fmaxf(m, mo);
  s = s * __expf(m - nm) + so * __expf(mo - nm);
  m = nm;
}

// ---------------------------------------------------------------------------
// convert f32 (rows x K) -> bf16 (rowsPad x K), zero-filling pad rows
// ---------------------------------------------------------------------------
__global__ void cvt_pad_kernel(const float* __restrict__ src, ushort_t* __restrict__ dst,
                               int rows, int rowsPad, int K) {
  size_t idx = (size_t)blockIdx.x * blockDim.x + threadIdx.x;
  size_t total = (size_t)rowsPad * (size_t)K;
  if (idx >= total) return;
  int row = (int)(idx / (size_t)K);
  dst[idx] = (row < rows) ? f2bf(src[idx]) : (ushort_t)0;
}

// ---------------------------------------------------------------------------
// cooperative stage of a 16 x K bf16 A panel into padded LDS.
// Uses the CDNA5 async global->LDS data path (tracked by ASYNCcnt) when
// enabled: per-lane 16B transfers, LDS byte address = low 32 bits of the flat
// shared-memory pointer (ISA 10.2 aperture).
// ---------------------------------------------------------------------------
template<int K, int SA>
__device__ __forceinline__ void stage_A(const ushort_t* __restrict__ A, int m0,
                                        ushort_t* sA, int tid) {
  constexpr int CH = K / 8;            // 16-byte chunks per row
#if USE_ASYNC_STAGE
  for (int i = tid; i < 16 * CH; i += 256) {
    int row = i / CH, c = i % CH;
    unsigned lds = (unsigned)(uintptr_t)&sA[row * SA + c * 8];
    unsigned long long g =
        (unsigned long long)(uintptr_t)(A + (size_t)m0 * K + (size_t)i * 8);
    asm volatile("global_load_async_to_lds_b128 %0, %1, off"
                 :: "v"(lds), "v"(g) : "memory");
  }
  asm volatile("s_wait_asynccnt 0x0" ::: "memory");
#else
  const uint4* src = (const uint4*)(A + (size_t)m0 * K);
  for (int i = tid; i < 16 * CH; i += 256) {
    int row = i / CH, c = i % CH;
    *(uint4*)&sA[row * SA + c * 8] = src[i];
  }
#endif
}

// ---------------------------------------------------------------------------
// Fragment layouts per CDNA5 ISA 7.12.2 (wave32, 16-bit operands):
//   A: lane<16 row=lane,  elems {k0..k0+7, k0+16..k0+23}
//      lane>=16 row=l-16, elems {k0+8..k0+15, k0+24..k0+31}
//   B: lane<16 col=lane,  elems {k0..k0+7, k0+16..k0+23} ; lane>=16 col=l-16, elems {k0+8..k0+15, k0+24..k0+31}
// ---------------------------------------------------------------------------
__device__ __forceinline__ v16bf load_afrag(const ushort_t* aRow, int k0) {
  return make_frag(*(const uint4*)(aRow + k0), *(const uint4*)(aRow + k0 + 16));
}
__device__ __forceinline__ v16bf load_bfrag(const ushort_t* wCol, int k0) {
  return make_frag(*(const uint4*)(wCol + k0), *(const uint4*)(wCol + k0 + 16));
}

// ---------------------------------------------------------------------------
// bottleneck GEMM: out(4096 x N, bf16) = A(4096 x K, bf16) @ W^T(N x K, bf16)
// ---------------------------------------------------------------------------
template<int K, int N>
__global__ __launch_bounds__(256)
void proj_gemm_kernel(const ushort_t* __restrict__ A, const ushort_t* __restrict__ W,
                      ushort_t* __restrict__ out) {
  constexpr int SA = K + 8;
  __shared__ __align__(16) ushort_t sA[16 * SA];
  const int tid = threadIdx.x, wave = tid >> 5, lane = tid & 31;
  const int half = lane >> 4, l15 = lane & 15;
  const int m0 = blockIdx.x * 16;

  stage_A<K, SA>(A, m0, sA, tid);
  __syncthreads();

  const ushort_t* aRow = &sA[l15 * SA + half * 8];
  for (int t = wave; t < N / 16; t += 8) {
    const int n0 = t * 16;
    const ushort_t* wCol = W + (size_t)(n0 + l15) * K + half * 8;
    v8f acc = {};
#pragma unroll 4
    for (int k0 = 0; k0 < K; k0 += 32)
      acc = __builtin_amdgcn_wmma_f32_16x16x32_bf16(
          false, load_afrag(aRow, k0), false, load_bfrag(wCol, k0),
          (short)0, acc, false, false);
#pragma unroll
    for (int r = 0; r < 8; ++r)
      out[(size_t)(m0 + half * 8 + r) * N + n0 + l15] = f2bf(acc[r]);
  }
}

// ---------------------------------------------------------------------------
// fused GEMM + online logsumexp over NPAD (nReal valid) classes.
// AREG: keep the wave's 16xK A panel resident in VGPRs (K<=256).
// HEAD: additionally gather fixed gate columns 1000/1001.
// ---------------------------------------------------------------------------
template<int K, int NPAD, bool HEAD, bool AREG>
__global__ __launch_bounds__(256)
void lse_stage_kernel(const ushort_t* __restrict__ A, const ushort_t* __restrict__ W,
                      int nReal, const int* __restrict__ target,
                      int clsBase, int clsEnd,
                      float* __restrict__ lse_out, float* __restrict__ tgt_out,
                      float* __restrict__ c0_out, float* __restrict__ c1_out) {
  constexpr int SA = K + 8;
  constexpr int KT = K / 32;
  __shared__ __align__(16) ushort_t sA[16 * SA];
  __shared__ float red_m[8][16];
  __shared__ float red_s[8][16];

  const int tid = threadIdx.x, wave = tid >> 5, lane = tid & 31;
  const int half = lane >> 4, l15 = lane & 15;
  const int m0 = blockIdx.x * 16;

  stage_A<K, SA>(A, m0, sA, tid);
  __syncthreads();

  const ushort_t* aRow = &sA[l15 * SA + half * 8];
  v16bf aF[AREG ? KT : 1];
  if constexpr (AREG) {
#pragma unroll
    for (int kt = 0; kt < KT; ++kt) aF[kt] = load_afrag(aRow, kt * 32);
  }

  // target class per row this lane's C values belong to (rows half*8 + r)
  int want[8];
#pragma unroll
  for (int r = 0; r < 8; ++r) {
    int t = target[m0 + half * 8 + r];
    want[r] = (t >= clsBase && t < clsEnd) ? (t - clsBase) : -1;
  }

  // per-lane running (max, sumexp) over this lane's column subset
  float m_run[8], s_run[8];
#pragma unroll
  for (int r = 0; r < 8; ++r) { m_run[r] = LSE_NEG; s_run[r] = 0.f; }

  for (int t = wave; t < NPAD / 16; t += 8) {
    const int n0 = t * 16;
    const int col = n0 + l15;
    const ushort_t* wCol = W + (size_t)col * K + half * 8;
    if (t + 8 < NPAD / 16)
      __builtin_prefetch(W + (size_t)(col + 128) * K, 0, 1);   // next B tile
    v8f acc = {};
#pragma unroll
    for (int kt = 0; kt < KT; ++kt) {
      v16bf a = AREG ? aF[kt] : load_afrag(aRow, kt * 32);
      acc = __builtin_amdgcn_wmma_f32_16x16x32_bf16(
          false, a, false, load_bfrag(wCol, kt * 32), (short)0, acc, false, false);
    }
    const bool colValid = (col < nReal);
#pragma unroll
    for (int r = 0; r < 8; ++r) {
      const float v = colValid ? acc[r] : LSE_NEG;   // mask padded classes
      const int row = m0 + half * 8 + r;
      if (want[r] == col) tgt_out[row] = v;          // want<nReal => col valid
      if constexpr (HEAD) {
        if (col == 1000) c0_out[row] = v;
        if (col == 1001) c1_out[row] = v;
      }
      float nm = fmaxf(m_run[r], v);
      s_run[r] = s_run[r] * __expf(m_run[r] - nm) + __expf(v - nm);
      m_run[r] = nm;
    }
  }

  // one butterfly merge across the 16-lane half group (xor bits 0..3)
#pragma unroll
  for (int r = 0; r < 8; ++r) {
    float m = m_run[r], s = s_run[r];
    for (int off = 8; off >= 1; off >>= 1) {
      float mo = __shfl_xor(m, off, 32);
      float so = __shfl_xor(s, off, 32);
      lse_merge(m, s, mo, so);
    }
    m_run[r] = m; s_run[r] = s;
  }
  if (l15 == 0) {                    // lanes 0 / 16 hold rows 0-7 / 8-15
#pragma unroll
    for (int r = 0; r < 8; ++r) {
      red_m[wave][half * 8 + r] = m_run[r];
      red_s[wave][half * 8 + r] = s_run[r];
    }
  }
  __syncthreads();
  if (tid < 16) {                    // merge the 8 wave partials per row
    float M = red_m[0][tid], S = red_s[0][tid];
    for (int w = 1; w < 8; ++w) lse_merge(M, S, red_m[w][tid], red_s[w][tid]);
    lse_out[m0 + tid] = M + __logf(S);
  }
}

// ---------------------------------------------------------------------------
// assemble per-token log prob + loss (single block, deterministic)
// ---------------------------------------------------------------------------
__global__ __launch_bounds__(256)
void finalize_kernel(const int* __restrict__ target,
                     const float* __restrict__ lse_head, const float* __restrict__ tgt_head,
                     const float* __restrict__ c1000, const float* __restrict__ c1001,
                     const float* __restrict__ lse_t0, const float* __restrict__ tgt_t0,
                     const float* __restrict__ lse_t1, const float* __restrict__ tgt_t1,
                     float* __restrict__ out) {
  __shared__ float red[256];
  float local = 0.f;
  for (int i = threadIdx.x; i < N_TOKENS; i += 256) {
    const int t = target[i];
    const float hl = lse_head[i];
    float o;
    if (t < 1000)       o = tgt_head[i] - hl;
    else if (t < 10000) o = (tgt_t0[i] - lse_t0[i]) + (c1000[i] - hl);
    else                o = (tgt_t1[i] - lse_t1[i]) + (c1001[i] - hl);
    out[i] = o;
    local += o;
  }
  red[threadIdx.x] = local;
  __syncthreads();
  for (int s = 128; s > 0; s >>= 1) {
    if (threadIdx.x < s) red[threadIdx.x] += red[threadIdx.x + s];
    __syncthreads();
  }
  if (threadIdx.x == 0) out[N_TOKENS] = -red[0] / (float)N_TOKENS;
}

// ---------------------------------------------------------------------------
extern "C" void kernel_launch(void* const* d_in, const int* in_sizes, int n_in,
                              void* d_out, int out_size, void* d_ws, size_t ws_size,
                              hipStream_t stream) {
  const float* x      = (const float*)d_in[0];
  const int*   target = (const int*)  d_in[1];
  const float* W_head = (const float*)d_in[2];
  const float* W0a    = (const float*)d_in[3];
  const float* W0b    = (const float*)d_in[4];
  const float* W1a    = (const float*)d_in[5];
  const float* W1b    = (const float*)d_in[6];
  float* out = (float*)d_out;

  // workspace layout
  char* ws = (char*)d_ws;
  auto alloc = [&](size_t bytes) -> char* {
    char* p = ws; ws += (bytes + 255) & ~(size_t)255; return p;
  };
  ushort_t* xb   = (ushort_t*)alloc((size_t)N_TOKENS * D_MODEL * 2);
  ushort_t* whb  = (ushort_t*)alloc((size_t)NH_PAD * D_MODEL * 2);
  ushort_t* w0ab = (ushort_t*)alloc((size_t)256 * D_MODEL * 2);
  ushort_t* w0bb = (ushort_t*)alloc((size_t)N0_PAD * 256 * 2);
  ushort_t* w1ab = (ushort_t*)alloc((size_t)64 * D_MODEL * 2);
  ushort_t* w1bb = (ushort_t*)alloc((size_t)N1_PAD * 64 * 2);
  ushort_t* p0b  = (ushort_t*)alloc((size_t)N_TOKENS * 256 * 2);
  ushort_t* p1b  = (ushort_t*)alloc((size_t)N_TOKENS * 64 * 2);
  float* lse_head = (float*)alloc(N_TOKENS * 4);
  float* tgt_head = (float*)alloc(N_TOKENS * 4);
  float* c1000    = (float*)alloc(N_TOKENS * 4);
  float* c1001    = (float*)alloc(N_TOKENS * 4);
  float* lse_t0   = (float*)alloc(N_TOKENS * 4);
  float* tgt_t0   = (float*)alloc(N_TOKENS * 4);
  float* lse_t1   = (float*)alloc(N_TOKENS * 4);
  float* tgt_t1   = (float*)alloc(N_TOKENS * 4);

  auto cvt = [&](const float* src, ushort_t* dst, int rows, int rowsPad, int K) {
    size_t total = (size_t)rowsPad * K;
    int grid = (int)((total + 255) / 256);
    cvt_pad_kernel<<<grid, 256, 0, stream>>>(src, dst, rows, rowsPad, K);
  };
  cvt(x,      xb,   N_TOKENS, N_TOKENS, D_MODEL);
  cvt(W_head, whb,  NH_REAL,  NH_PAD,   D_MODEL);
  cvt(W0a,    w0ab, 256,      256,      D_MODEL);
  cvt(W0b,    w0bb, N0_REAL,  N0_PAD,   256);
  cvt(W1a,    w1ab, 64,       64,       D_MODEL);
  cvt(W1b,    w1bb, N1_REAL,  N1_PAD,   64);

  const int rowBlocks = N_TOKENS / 16;   // 256 workgroups

  proj_gemm_kernel<D_MODEL, 256><<<rowBlocks, 256, 0, stream>>>(xb, w0ab, p0b);
  proj_gemm_kernel<D_MODEL, 64 ><<<rowBlocks, 256, 0, stream>>>(xb, w1ab, p1b);

  lse_stage_kernel<D_MODEL, NH_PAD, true, false><<<rowBlocks, 256, 0, stream>>>(
      xb, whb, NH_REAL, target, 0, 1000, lse_head, tgt_head, c1000, c1001);
  lse_stage_kernel<256, N0_PAD, false, true><<<rowBlocks, 256, 0, stream>>>(
      p0b, w0bb, N0_REAL, target, 1000, 10000, lse_t0, tgt_t0, nullptr, nullptr);
  lse_stage_kernel<64, N1_PAD, false, true><<<rowBlocks, 256, 0, stream>>>(
      p1b, w1bb, N1_REAL, target, 10000, 50257, lse_t1, tgt_t1, nullptr, nullptr);

  finalize_kernel<<<1, 256, 0, stream>>>(
      target, lse_head, tgt_head, c1000, c1001,
      lse_t0, tgt_t0, lse_t1, tgt_t1, out);
}
